// ExportEquiGraspSO3DeformableAttn2_31044023615891
// MI455X (gfx1250) — matrix operand order repaired and checked by reading, not compiled, not run on hardware
//
#include <hip/hip_runtime.h>

typedef _Float16 half_t;
typedef __attribute__((ext_vector_type(16))) _Float16 v16h;
typedef __attribute__((ext_vector_type(8)))  float    v8f;

constexpr int BS = 8, NS = 2048, NCP = 8, NH = 8, C = 128, HID = 128, R = 128;
constexpr int TOK = 16;        // tokens per workgroup (one WMMA M-tile)
constexpr int THREADS = 256;   // 8 wave32s
constexpr int KT = C / 32;     // 4 K-tiles per GEMM
constexpr int SWZ = 40;        // halves per (ktile, n) record: 32 data + 8 pad (80B, 16B-aligned)

// ---- WMMA fragment loaders (CDNA5 16x16x32 f16 layouts, wave32) ----
// A (16x32, MxK): lanes 0-15 hold M=lane, halves 0..15 = K(k0..k0+15);
// lanes 16-31 hold K(k0+16..k0+31).
__device__ inline v16h load_a_frag(const half_t* __restrict__ A, int ldk,
                                   int row, int k0, int half_sel) {
  const half_t* p = A + row * ldk + k0 + half_sel * 16;
  v16h r;
#pragma unroll
  for (int i = 0; i < 8; ++i) r[i] = p[i];
#pragma unroll
  for (int i = 0; i < 8; ++i) r[8 + i] = p[8 + i];
  return r;
}
// B from swizzled LDS: record (kt, n) holds B[kt*32 .. kt*32+31][n] contiguous.
// Lane (half_sel, col): halves = K(kt*32 + half_sel*16 + 0..15) -> 32 contiguous
// bytes at a 16B-aligned address (two ds_load_b128, conflict-free stride).
__device__ inline v16h load_b_swz(const half_t* __restrict__ Bs, int kt,
                                  int col, int half_sel) {
  const half_t* p = Bs + (kt * 128 + col) * SWZ + half_sel * 16;
  v16h r;
#pragma unroll
  for (int i = 0; i < 16; ++i) r[i] = p[i];
  return r;
}
// Stage a (K=128 x N=128) f32 weight matrix into swizzled f16 LDS layout.
__device__ inline void stage_weights_swz(half_t* __restrict__ Bs,
                                         const float* __restrict__ W, int tid) {
  for (int i = tid; i < C * HID; i += THREADS) {
    int k = i >> 7, n = i & 127;
    Bs[((k >> 5) * 128 + n) * SWZ + (k & 31)] = (half_t)W[i];
  }
}

// Bilinear triplane sample of 8 consecutive channels, accumulated with weight.
__device__ inline void sample_accum(const float* __restrict__ pl0,
                                    const float* __restrict__ pl1,
                                    const float* __restrict__ pl2,
                                    const float* __restrict__ pos,
                                    int cbase, float* __restrict__ acc,
                                    float wgt) {
  const float P0 = pos[0], P1 = pos[1], P2 = pos[2];
  const float* planes[3] = {pl0, pl1, pl2};
  const float U[3] = {P0, P0, P1};   // (x,z) (x,y) (y,z) per reference
  const float V[3] = {P2, P1, P2};
#pragma unroll
  for (int pl = 0; pl < 3; ++pl) {
    float x = fminf(fmaxf(U[pl], 0.f), 1.f) * (float)(R - 1);
    float y = fminf(fmaxf(V[pl], 0.f), 1.f) * (float)(R - 1);
    float x0f = floorf(x), y0f = floorf(y);
    float wx = x - x0f, wy = y - y0f;
    int x0 = (int)x0f, y0 = (int)y0f;
    int x1 = (x0 + 1 < R - 1) ? x0 + 1 : R - 1;
    int y1 = (y0 + 1 < R - 1) ? y0 + 1 : R - 1;
    float w00 = (1.f - wx) * (1.f - wy), w01 = wx * (1.f - wy);
    float w10 = (1.f - wx) * wy,         w11 = wx * wy;
    const float* base = planes[pl];
#pragma unroll
    for (int r = 0; r < 8; ++r) {
      const float* chan = base + (size_t)(cbase + r) * (R * R);
      float f00 = chan[y0 * R + x0];
      float f01 = chan[y0 * R + x1];
      float f10 = chan[y1 * R + x0];
      float f11 = chan[y1 * R + x1];
      acc[r] += wgt * (f00 * w00 + f01 * w01 + f10 * w10 + f11 * w11);
    }
  }
}

__global__ __launch_bounds__(THREADS) void equigrasp_fused(
    const float* __restrict__ qp,  const float* __restrict__ pxz,
    const float* __restrict__ pxy, const float* __restrict__ pyz,
    const float* __restrict__ cp,  const float* __restrict__ Wv,
    const float* __restrict__ bv,  const float* __restrict__ Ww,
    const float* __restrict__ bw,  const float* __restrict__ Wo,
    const float* __restrict__ bo,  float* __restrict__ out) {
  __shared__ __align__(16) half_t WvS[KT * 128 * SWZ];   // 40 KB swizzled W_v
  __shared__ __align__(16) half_t WoS[KT * 128 * SWZ];   // 40 KB swizzled W_o
  __shared__ __align__(16) half_t As[TOK * C];           // weighted anchor feature
  __shared__ __align__(16) half_t Hs[TOK * HID];         // hidden
  __shared__ float featL[TOK][C];
  __shared__ float posL[TOK][9][3];
  __shared__ float WwsL[C][NH];                          // W_w collapsed over NCP
  __shared__ float wsumL[TOK][NH];
  __shared__ float wsumsumL[TOK];
  __shared__ float bvL[HID], boL[C], bwsL[NH];

  const int tid = threadIdx.x;
  const int wg  = blockIdx.x;
  const int b   = wg >> 7;              // 128 workgroups per batch
  const int t0  = (wg & 127) * TOK;

  const size_t pstride = (size_t)C * R * R;
  const float* pl0 = pxz + (size_t)b * pstride;
  const float* pl1 = pxy + (size_t)b * pstride;
  const float* pl2 = pyz + (size_t)b * pstride;

  // Stage both weight matrices (f32 -> f16, fragment-major swizzle).
  stage_weights_swz(WvS, Wv, tid);
  stage_weights_swz(WoS, Wo, tid);
  for (int i = tid; i < C * NH; i += THREADS) {
    int c = i >> 3, j = i & 7;
    float s = 0.f;
#pragma unroll
    for (int g = 0; g < NCP; ++g) s += Ww[c * (NCP * NH) + g * NH + j];
    WwsL[c][j] = s;
  }
  if (tid < NH) {
    float s = 0.f;
#pragma unroll
    for (int g = 0; g < NCP; ++g) s += bw[g * NH + tid];
    bwsL[tid] = s;
  }
  if (tid < HID) bvL[tid] = bv[tid];
  else           boL[tid - HID] = bo[tid - HID];

  // Per-token geometry: rotation from 6D + anchor positions.
  if (tid < TOK) {
    const float* q = qp + ((size_t)b * NS + t0 + tid) * 9;
    float px = q[0], py = q[1], pz = q[2];
    float a1x = q[3], a1y = q[4], a1z = q[5];
    float a2x = q[6], a2y = q[7], a2z = q[8];
    float inv = rsqrtf(a1x * a1x + a1y * a1y + a1z * a1z);
    float b1x = a1x * inv, b1y = a1y * inv, b1z = a1z * inv;
    float d = b1x * a2x + b1y * a2y + b1z * a2z;
    float b2x = a2x - d * b1x, b2y = a2y - d * b1y, b2z = a2z - d * b1z;
    float inv2 = rsqrtf(b2x * b2x + b2y * b2y + b2z * b2z);
    b2x *= inv2; b2y *= inv2; b2z *= inv2;
    float b3x = b1y * b2z - b1z * b2y;
    float b3y = b1z * b2x - b1x * b2z;
    float b3z = b1x * b2y - b1y * b2x;
    posL[tid][0][0] = px; posL[tid][0][1] = py; posL[tid][0][2] = pz;
#pragma unroll
    for (int g = 0; g < NCP; ++g) {
      float cx = cp[g * 3 + 0], cy = cp[g * 3 + 1], cz = cp[g * 3 + 2];
      posL[tid][1 + g][0] = px + b1x * cx + b1y * cy + b1z * cz;
      posL[tid][1 + g][1] = py + b2x * cx + b2y * cy + b2z * cz;
      posL[tid][1 + g][2] = pz + b3x * cx + b3y * cy + b3z * cz;
    }
  }
  __syncthreads();

  const int t = tid >> 4, lt = tid & 15, cbase = lt * 8;

  // feat = triplane(query)
  float facc[8] = {0, 0, 0, 0, 0, 0, 0, 0};
  sample_accum(pl0, pl1, pl2, &posL[t][0][0], cbase, facc, 1.0f);
#pragma unroll
  for (int r = 0; r < 8; ++r) featL[t][cbase + r] = facc[r];
  __syncthreads();

  // wsum[t][j] = feat . Wws[:,j] + bws[j]  (collapsed over NCP axis)
  if (tid < TOK * NH) {
    int tt = tid >> 3, j = tid & 7;
    float s = bwsL[j];
    for (int c = 0; c < C; ++c) s += featL[tt][c] * WwsL[c][j];
    wsumL[tt][j] = s;
  }
  __syncthreads();
  if (tid < TOK) {
    float s = 0.f;
#pragma unroll
    for (int j = 0; j < NH; ++j) s += wsumL[tid][j];
    wsumsumL[tid] = s;
  }
  __syncthreads();

  // s[t][c] = sum_g wsum[g] * triplane(anchor_g)[c]   (sf never materialized)
  float sacc[8] = {0, 0, 0, 0, 0, 0, 0, 0};
#pragma unroll
  for (int g = 0; g < NCP; ++g)
    sample_accum(pl0, pl1, pl2, &posL[t][1 + g][0], cbase, sacc, wsumL[t][g]);
#pragma unroll
  for (int r = 0; r < 8; ++r) As[t * C + cbase + r] = (half_t)sacc[r];
  __syncthreads();

  // ---- GEMM 1: h = s @ W_v + (sum wsum) * b_v  (M=16,N=128,K=128, WMMA) ----
  const int wv = tid >> 5, lane = tid & 31;
  const int half_sel = lane >> 4, lm = lane & 15;
  const int n0 = wv * 16, col = n0 + lm;
  v8f acc = {};
#pragma unroll
  for (int kt = 0; kt < KT; ++kt) {
    v16h af = load_a_frag(As, C, lm, kt * 32, half_sel);
    v16h bf = load_b_swz(WvS, kt, col, half_sel);
    acc = __builtin_amdgcn_wmma_f32_16x16x32_f16(false, af, false, bf,
                                                 (short)0, acc, false, false);
  }
#pragma unroll
  for (int r = 0; r < 8; ++r) {
    int trow = half_sel * 8 + r;
    float h = acc[r] + wsumsumL[trow] * bvL[col];
    Hs[trow * HID + col] = (half_t)h;
  }
  __syncthreads();

  // ---- GEMM 2: out = h @ W_o + b_o + feat ----
  v8f acc2 = {};
#pragma unroll
  for (int kt = 0; kt < KT; ++kt) {
    v16h af = load_a_frag(Hs, HID, lm, kt * 32, half_sel);
    v16h bf = load_b_swz(WoS, kt, col, half_sel);
    acc2 = __builtin_amdgcn_wmma_f32_16x16x32_f16(false, af, false, bf,
                                                  (short)0, acc2, false, false);
  }
#pragma unroll
  for (int r = 0; r < 8; ++r) {
    int trow = half_sel * 8 + r;
    float o = acc2[r] + boL[col] + featL[trow][col];
    out[((size_t)b * NS + t0 + trow) * C + col] = o;
  }
}

extern "C" void kernel_launch(void* const* d_in, const int* in_sizes, int n_in,
                              void* d_out, int out_size, void* d_ws, size_t ws_size,
                              hipStream_t stream) {
  (void)in_sizes; (void)n_in; (void)out_size; (void)d_ws; (void)ws_size;
  const float* qp  = (const float*)d_in[0];
  const float* pxz = (const float*)d_in[1];
  const float* pxy = (const float*)d_in[2];
  const float* pyz = (const float*)d_in[3];
  const float* cp  = (const float*)d_in[4];
  const float* Wv  = (const float*)d_in[5];
  const float* bv  = (const float*)d_in[6];
  const float* Ww  = (const float*)d_in[7];
  const float* bw  = (const float*)d_in[8];
  const float* Wo  = (const float*)d_in[9];
  const float* bo  = (const float*)d_in[10];
  float* out = (float*)d_out;
  dim3 grid(BS * NS / TOK);   // 1024 workgroups, 16 tokens each
  equigrasp_fused<<<grid, THREADS, 0, stream>>>(qp, pxz, pxy, pyz, cp, Wv, bv,
                                                Ww, bw, Wo, bo, out);
}
